// HierarchicalMemory_81020263071870
// MI455X (gfx1250) — matrix operand order repaired and checked by reading, not compiled, not run on hardware
//
#include <hip/hip_runtime.h>

typedef __attribute__((ext_vector_type(16))) _Float16 v16h;
typedef __attribute__((ext_vector_type(8)))  _Float16 v8h;
typedef __attribute__((ext_vector_type(8)))  float    v8f;
typedef __attribute__((ext_vector_type(4)))  unsigned int u32x4;
typedef __attribute__((ext_vector_type(8)))  unsigned int u32x8;

#define HDIM 512
#define MDIM 4096
#define NTOK 16384
#define INVT 10.0f   // 1/TEMP

// ---------------------------------------------------------------- WMMA utils

__device__ __forceinline__ v8f wmma32(v16h a, v16h b, v8f c) {
  return __builtin_amdgcn_wmma_f32_16x16x32_f16(false, a, false, b, (short)0, c,
                                                false, false);
}

// A-fragment (16x32, MxK, f16). Row-major source, row stride `ld` halves.
__device__ __forceinline__ v16h load_fragA(const _Float16* base, int ld) {
  const int lane = (int)(threadIdx.x & 31u);
  const int row  = lane & 15;
  const int koff = (lane >> 4) << 3;        // 0 or 8
  const _Float16* p = base + row * ld + koff;
  v8h lo = *(const v8h*)(p);
  v8h hi = *(const v8h*)(p + 16);
  return __builtin_shufflevector(lo, hi, 0,1,2,3,4,5,6,7,8,9,10,11,12,13,14,15);
}

// B-fragment (32x16, KxN, f16) from a row-major [N][K] source (B^T), stride `ld`.
__device__ __forceinline__ v16h load_fragB(const _Float16* base, int ld) {
  const int lane = (int)(threadIdx.x & 31u);
  const int col  = lane & 15;
  const int koff = (lane >> 4) << 3;        // 0 or 8
  const _Float16* p = base + col * ld + koff;
  v8h lo = *(const v8h*)(p);
  v8h hi = *(const v8h*)(p + 16);
  return __builtin_shufflevector(lo, hi, 0,1,2,3,4,5,6,7,8,9,10,11,12,13,14,15);
}

__device__ __forceinline__ float half_max(float v) {
  #pragma unroll
  for (int m = 1; m <= 8; m <<= 1) v = fmaxf(v, __shfl_xor(v, m, 32));
  return v;
}
__device__ __forceinline__ float half_sum(float v) {
  #pragma unroll
  for (int m = 1; m <= 8; m <<= 1) v += __shfl_xor(v, m, 32);
  return v;
}

// ---------------------------------------------------------------- TDM (Tensor Data Mover)
// Issue one 2-D tensor_load_to_lds per tile: tile_d1 rows of tile_d0 f16
// elements, source row pitch stride0 (elements). D# layout per CDNA5 ISA
// CDNA5 ISA (data_size=2 bytes, count=1, type=2).

__device__ __forceinline__ void tdm_load_2d_f16(unsigned lds_addr,
                                                const void* gaddr,
                                                unsigned tensor_d0,
                                                unsigned tensor_d1,
                                                unsigned long long stride0,
                                                unsigned tile_d0,
                                                unsigned tile_d1) {
  unsigned long long ga = (unsigned long long)(size_t)gaddr;
  u32x4 g0;
  g0[0] = 1u;                                              // count=1 (valid user D#)
  g0[1] = lds_addr;                                        // LDS byte address
  g0[2] = (unsigned)(ga & 0xffffffffu);                    // global_addr[31:0]
  g0[3] = (unsigned)((ga >> 32) & 0x1ffffffu) | (2u << 30);// addr[56:32] | type=2
  u32x8 g1;
  g1[0] = (1u << 16);                                      // wg_mask=0, data_size=1 (2B)
  g1[1] = (tensor_d0 & 0xffffu) << 16;                     // tensor_dim0[15:0]
  g1[2] = ((tensor_d0 >> 16) & 0xffffu) |                  // tensor_dim0[31:16]
          ((tensor_d1 & 0xffffu) << 16);                   // tensor_dim1[15:0]
  g1[3] = ((tensor_d1 >> 16) & 0xffffu) |                  // tensor_dim1[31:16]
          ((tile_d0 & 0xffffu) << 16);                     // tile_dim0
  g1[4] = (tile_d1 & 0xffffu);                             // tile_dim1 (tile_dim2=0)
  g1[5] = (unsigned)(stride0 & 0xffffffffu);               // tensor_dim0_stride lo
  g1[6] = (unsigned)((stride0 >> 32) & 0xffffu);           // stride hi (dim1_stride=0)
  g1[7] = 0u;
  asm volatile("tensor_load_to_lds %0, %1" :: "s"(g0), "s"(g1) : "memory");
}

// ---------------------------------------------------------------- prep kernels

__global__ void k_f32_to_f16(const float* __restrict__ src,
                             _Float16* __restrict__ dst, int n) {
  int i = blockIdx.x * blockDim.x + threadIdx.x;
  if (i < n) dst[i] = (_Float16)src[i];
}

__global__ void k_transpose_f16(const float* __restrict__ V,
                                _Float16* __restrict__ Vt) {
  int i = blockIdx.x * blockDim.x + threadIdx.x;
  if (i < MDIM * HDIM) {
    int m = i / HDIM, h = i - m * HDIM;
    Vt[h * MDIM + m] = (_Float16)V[i];
  }
}

__global__ void k_norm_keys(const float* __restrict__ K,
                            _Float16* __restrict__ K16) {
  int row  = blockIdx.x * (blockDim.x >> 5) + (threadIdx.x >> 5);
  int lane = threadIdx.x & 31;
  const float* r = K + (size_t)row * HDIM;
  float ss = 0.f;
  #pragma unroll
  for (int i = 0; i < HDIM / 32; i++) { float v = r[lane + 32 * i]; ss += v * v; }
  #pragma unroll
  for (int m = 1; m <= 16; m <<= 1) ss += __shfl_xor(ss, m, 32);
  float sc = rsqrtf(ss + 1e-6f);
  #pragma unroll
  for (int i = 0; i < HDIM / 32; i++)
    K16[(size_t)row * HDIM + lane + 32 * i] = (_Float16)(r[lane + 32 * i] * sc);
}

__global__ void k_mask(const int* __restrict__ usage, float* __restrict__ maskF) {
  int i = blockIdx.x * blockDim.x + threadIdx.x;
  if (i < MDIM) maskF[i] = usage[i] > 0 ? 1.0f : 0.0f;
}

// ---------------------------------------------------------------- GEMM C = A*B^T (+bias)

template <bool NORM>
__global__ __launch_bounds__(256)
void k_gemm_rt(const _Float16* __restrict__ A, const _Float16* __restrict__ Bw,
               const float* __restrict__ bias, _Float16* __restrict__ outH,
               float* __restrict__ outF) {
  const int wave = threadIdx.x >> 5;
  const int lane = threadIdx.x & 31;
  const int r0   = blockIdx.x * 128 + wave * 16;
  const int drow = (lane >> 4) << 3;
  const int dcol = lane & 15;

  v16h af[16];
  #pragma unroll
  for (int kt = 0; kt < 16; kt++)
    af[kt] = load_fragA(A + (size_t)r0 * HDIM + kt * 32, HDIM);

  v8f acc[32];
  const v8f zv = {};
  #pragma unroll
  for (int nt = 0; nt < 32; nt++) {
    v8f c = zv;
    #pragma unroll
    for (int kt = 0; kt < 16; kt++)
      c = wmma32(af[kt], load_fragB(Bw + (size_t)(nt * 16) * HDIM + kt * 32, HDIM), c);
    float b = bias[nt * 16 + dcol];
    #pragma unroll
    for (int j = 0; j < 8; j++) c[j] += b;
    if constexpr (NORM) {
      acc[nt] = c;
    } else {
      #pragma unroll
      for (int j = 0; j < 8; j++)
        outF[(size_t)(r0 + drow + j) * HDIM + nt * 16 + dcol] = c[j];
    }
  }

  if constexpr (NORM) {
    float ss[8];
    #pragma unroll
    for (int j = 0; j < 8; j++) ss[j] = 0.f;
    #pragma unroll
    for (int nt = 0; nt < 32; nt++)
      #pragma unroll
      for (int j = 0; j < 8; j++) ss[j] += acc[nt][j] * acc[nt][j];
    float rs[8];
    #pragma unroll
    for (int j = 0; j < 8; j++) rs[j] = rsqrtf(half_sum(ss[j]) + 1e-6f);
    #pragma unroll
    for (int nt = 0; nt < 32; nt++)
      #pragma unroll
      for (int j = 0; j < 8; j++)
        outH[(size_t)(r0 + drow + j) * HDIM + nt * 16 + dcol] =
            (_Float16)(acc[nt][j] * rs[j]);
  }
}

// ---------------------------------------------------------------- fused attention

__global__ __launch_bounds__(256)
void k_attention(const _Float16* __restrict__ Q,    // [NTOK,512] normalized f16
                 const _Float16* __restrict__ K16,  // [M,512]   normalized f16
                 const _Float16* __restrict__ Vt,   // [512,M]   f16 (transposed V)
                 const float* __restrict__ maskF,   // [M] 1/0
                 _Float16* __restrict__ R) {        // [NTOK,512] f16
  __shared__ __attribute__((aligned(32))) _Float16 sK[128 * 128];   // [m][h-chunk]
  __shared__ __attribute__((aligned(32))) _Float16 sV[128 * 128];   // [h-chunk][m]
  __shared__ __attribute__((aligned(32))) _Float16 sP[8][16 * 128]; // per-wave P

  const int wave = threadIdx.x >> 5;
  const int lane = threadIdx.x & 31;
  const int r0   = blockIdx.x * 128 + wave * 16;
  const int drow = (lane >> 4) << 3;
  const int dcol = lane & 15;
  const v8f zv = {};

  v16h qf[16];
  #pragma unroll
  for (int kt = 0; kt < 16; kt++)
    qf[kt] = load_fragA(Q + (size_t)r0 * HDIM + kt * 32, HDIM);

  v8f o[32];
  #pragma unroll
  for (int ot = 0; ot < 32; ot++) o[ot] = zv;
  float mrun[8], lrun[8];
  #pragma unroll
  for (int j = 0; j < 8; j++) { mrun[j] = -1e30f; lrun[j] = 0.f; }

  #pragma unroll 1
  for (int ct = 0; ct < 32; ct++) {
    const int m0 = ct * 128;

    // ---- scores S = Qn * Kn^T over H in 4 chunks of 128
    v8f s[8];
    #pragma unroll
    for (int nt = 0; nt < 8; nt++) s[nt] = zv;

    #pragma unroll
    for (int hc = 0; hc < 4; hc++) {
      __syncthreads();
      if (wave == 0) {
        // DMA K tile: 128 rows (m) x 128 halves (h) into sK via the TDM
        tdm_load_2d_f16((unsigned)(size_t)(void*)sK,
                        K16 + (size_t)m0 * HDIM + hc * 128,
                        HDIM, MDIM, HDIM, 128u, 128u);
        __builtin_amdgcn_s_wait_tensorcnt(0);
        if (ct + 1 < 32)  // global_prefetch_b8 of next K tile into L2
          __builtin_prefetch(K16 + (size_t)(m0 + 128) * HDIM + hc * 128, 0, 3);
      }
      __syncthreads();
      #pragma unroll
      for (int nt = 0; nt < 8; nt++) {
        v16h b0 = load_fragB(sK + nt * 16 * 128 + 0 * 32, 128);
        v16h b1 = load_fragB(sK + nt * 16 * 128 + 1 * 32, 128);
        v16h b2 = load_fragB(sK + nt * 16 * 128 + 2 * 32, 128);
        v16h b3 = load_fragB(sK + nt * 16 * 128 + 3 * 32, 128);
        s[nt] = wmma32(qf[hc * 4 + 0], b0, s[nt]);
        s[nt] = wmma32(qf[hc * 4 + 1], b1, s[nt]);
        s[nt] = wmma32(qf[hc * 4 + 2], b2, s[nt]);
        s[nt] = wmma32(qf[hc * 4 + 3], b3, s[nt]);
      }
    }

    // ---- temperature + usage mask + online softmax
    float mloc[8];
    #pragma unroll
    for (int j = 0; j < 8; j++) mloc[j] = -1e30f;
    #pragma unroll
    for (int nt = 0; nt < 8; nt++) {
      float mk = maskF[m0 + nt * 16 + dcol];
      #pragma unroll
      for (int j = 0; j < 8; j++) {
        float v = (mk > 0.5f) ? s[nt][j] * INVT : -1e9f;
        s[nt][j] = v;
        mloc[j]  = fmaxf(mloc[j], v);
      }
    }
    float mnew[8], alpha[8];
    #pragma unroll
    for (int j = 0; j < 8; j++) {
      float rm = half_max(mloc[j]);
      mnew[j]  = fmaxf(mrun[j], rm);
      alpha[j] = __expf(mrun[j] - mnew[j]);
      mrun[j]  = mnew[j];
    }
    float rsum[8];
    #pragma unroll
    for (int j = 0; j < 8; j++) rsum[j] = 0.f;
    #pragma unroll
    for (int nt = 0; nt < 8; nt++)
      #pragma unroll
      for (int j = 0; j < 8; j++) {
        float p = __expf(s[nt][j] - mnew[j]);
        rsum[j] += p;
        sP[wave][(drow + j) * 128 + nt * 16 + dcol] = (_Float16)p;
      }
    #pragma unroll
    for (int j = 0; j < 8; j++) lrun[j] = lrun[j] * alpha[j] + half_sum(rsum[j]);
    #pragma unroll
    for (int ot = 0; ot < 32; ot++)
      #pragma unroll
      for (int j = 0; j < 8; j++) o[ot][j] *= alpha[j];

    // ---- O += P * V
    v16h pf[4];
    #pragma unroll
    for (int kt = 0; kt < 4; kt++)
      pf[kt] = load_fragA(&sP[wave][kt * 32], 128);

    #pragma unroll
    for (int hc = 0; hc < 4; hc++) {
      __syncthreads();
      if (wave == 0) {
        // DMA V^T tile: 128 rows (h) x 128 halves (m) into sV via the TDM
        tdm_load_2d_f16((unsigned)(size_t)(void*)sV,
                        Vt + (size_t)(hc * 128) * MDIM + m0,
                        MDIM, HDIM, MDIM, 128u, 128u);
        __builtin_amdgcn_s_wait_tensorcnt(0);
      }
      __syncthreads();
      #pragma unroll
      for (int ol = 0; ol < 8; ol++) {
        const int ot = hc * 8 + ol;
        v16h b0 = load_fragB(sV + ol * 16 * 128 + 0 * 32, 128);
        v16h b1 = load_fragB(sV + ol * 16 * 128 + 1 * 32, 128);
        v16h b2 = load_fragB(sV + ol * 16 * 128 + 2 * 32, 128);
        v16h b3 = load_fragB(sV + ol * 16 * 128 + 3 * 32, 128);
        o[ot] = wmma32(pf[0], b0, o[ot]);
        o[ot] = wmma32(pf[1], b1, o[ot]);
        o[ot] = wmma32(pf[2], b2, o[ot]);
        o[ot] = wmma32(pf[3], b3, o[ot]);
      }
    }
  }

  // ---- finalize
  float inv[8];
  #pragma unroll
  for (int j = 0; j < 8; j++) inv[j] = 1.0f / lrun[j];
  #pragma unroll
  for (int ot = 0; ot < 32; ot++)
    #pragma unroll
    for (int j = 0; j < 8; j++)
      R[(size_t)(r0 + drow + j) * HDIM + ot * 16 + dcol] =
          (_Float16)(o[ot][j] * inv[j]);
}

// ---------------------------------------------------------------- launch

extern "C" void kernel_launch(void* const* d_in, const int* in_sizes, int n_in,
                              void* d_out, int out_size, void* d_ws, size_t ws_size,
                              hipStream_t stream) {
  const float* hidden = (const float*)d_in[0];
  const float* mkeys  = (const float*)d_in[1];
  const float* mvals  = (const float*)d_in[2];
  const float* Wk     = (const float*)d_in[3];
  const float* bk     = (const float*)d_in[4];
  const float* Wo     = (const float*)d_in[5];
  const float* bo     = (const float*)d_in[6];
  const int*   usage  = (const int*)d_in[7];

  char* ws = (char*)d_ws;
  _Float16* H16  = (_Float16*)ws; ws += (size_t)NTOK * HDIM * 2;
  _Float16* Q16  = (_Float16*)ws; ws += (size_t)NTOK * HDIM * 2;
  _Float16* R16  = (_Float16*)ws; ws += (size_t)NTOK * HDIM * 2;
  _Float16* K16  = (_Float16*)ws; ws += (size_t)MDIM * HDIM * 2;
  _Float16* Vt16 = (_Float16*)ws; ws += (size_t)MDIM * HDIM * 2;
  _Float16* Wk16 = (_Float16*)ws; ws += (size_t)HDIM * HDIM * 2;
  _Float16* Wo16 = (_Float16*)ws; ws += (size_t)HDIM * HDIM * 2;
  float*    maskF = (float*)ws;   ws += (size_t)MDIM * 4;

  const int nHid = NTOK * HDIM;
  const int nW   = HDIM * HDIM;

  k_f32_to_f16<<<(nHid + 255) / 256, 256, 0, stream>>>(hidden, H16, nHid);
  k_f32_to_f16<<<(nW + 255) / 256, 256, 0, stream>>>(Wk, Wk16, nW);
  k_f32_to_f16<<<(nW + 255) / 256, 256, 0, stream>>>(Wo, Wo16, nW);
  k_transpose_f16<<<(MDIM * HDIM + 255) / 256, 256, 0, stream>>>(mvals, Vt16);
  k_norm_keys<<<MDIM / 8, 256, 0, stream>>>(mkeys, K16);
  k_mask<<<MDIM / 256, 256, 0, stream>>>(usage, maskF);

  k_gemm_rt<true><<<NTOK / 128, 256, 0, stream>>>(H16, Wk16, bk, Q16, nullptr);
  k_attention<<<NTOK / 128, 256, 0, stream>>>(Q16, K16, Vt16, maskF, R16);
  k_gemm_rt<false><<<NTOK / 128, 256, 0, stream>>>(R16, Wo16, bo, nullptr,
                                                   (float*)d_out);
}
